// NumDecoder_72756745994932
// MI455X (gfx1250) — matrix operand order repaired and checked by reading, not compiled, not run on hardware
//
#include <hip/hip_runtime.h>
#include <stdint.h>

#define BB 64      // batch
#define TT 64      // seq len
#define VV 32000   // vocab
#define EE 256     // emb
#define HH 512     // hidden
#define XW (EE + HH)   // 768: concatenated [x | h] row width

typedef __attribute__((ext_vector_type(16))) __bf16 v16bf;
typedef __attribute__((ext_vector_type(8)))  float  v8f;
typedef __attribute__((ext_vector_type(2)))  float  v2f;

// ---------------- init / gather ----------------

__global__ void init_kernel(const float* __restrict__ eh, const float* __restrict__ ec,
                            float* __restrict__ xg0,   // [BB][XW], fill h part
                            float* __restrict__ cbuf,  // [BB][HH]
                            int* __restrict__ tokens) {
    int i = blockIdx.x * blockDim.x + threadIdx.x;
    if (i < BB * HH) {
        int b = i / HH, j = i % HH;
        xg0[b * XW + EE + j] = eh[i];
        cbuf[i] = ec[i];
    }
    if (i < BB) tokens[i] = 0;  // SOS
}

// xg[p][b][0:EE] = embedding[tokens[b]]
__global__ __launch_bounds__(256) void embed_kernel(const float* __restrict__ emb,
                                                    const int* __restrict__ tokens,
                                                    float* __restrict__ xg) {
    int b = blockIdx.x;           // 64 blocks
    int e = threadIdx.x;          // 256 threads
    xg[b * XW + e] = emb[(long)tokens[b] * EE + e];
}

// one-time fc_w f32 -> bf16 (used when ws_size permits)
__global__ __launch_bounds__(256) void convert_fcw_kernel(const float* __restrict__ w,
                                                          __bf16* __restrict__ wb) {
    long i = ((long)blockIdx.x * 256 + threadIdx.x) * 8;   // grid 8000 -> 16.384M elems
    float4 f0 = *(const float4*)(w + i);
    float4 f1 = *(const float4*)(w + i + 4);
    union { __bf16 h[8]; uint4 q; } o;
    o.h[0] = (__bf16)f0.x; o.h[1] = (__bf16)f0.y; o.h[2] = (__bf16)f0.z; o.h[3] = (__bf16)f0.w;
    o.h[4] = (__bf16)f1.x; o.h[5] = (__bf16)f1.y; o.h[6] = (__bf16)f1.z; o.h[7] = (__bf16)f1.w;
    *(uint4*)(wb + i) = o.q;
}

// ---------------- LSTM cell: full-fp32 WMMA (v_wmma_f32_16x16x4_f32) ----------------
// Wave owns a (16 batch x 16 hidden) tile and accumulates all 4 gates, then does the
// pointwise LSTM update in-register. Grid: 16 blocks x 8 waves = 4 batch tiles x 32 j-tiles.
__global__ __launch_bounds__(256) void cell_wmma_kernel(
    const float* __restrict__ xg_in,   // [BB][XW] = [x | h_prev]
    const float* __restrict__ w_ih,    // [4H][EE]
    const float* __restrict__ w_hh,    // [4H][HH]
    const float* __restrict__ b_ih, const float* __restrict__ b_hh,
    float* __restrict__ cbuf,          // [BB][HH], updated in place
    float* __restrict__ xg_out,        // [BB][XW]: h written to cols EE..EE+HH
    __bf16* __restrict__ hbf)          // [BB][HH]
{
    const int lane = threadIdx.x & 31;
    const int wave = threadIdx.x >> 5;
    const int m0 = (wave & 3) * 16;                    // batch tile
    const int n0 = (blockIdx.x * 2 + (wave >> 2)) * 16; // hidden tile
    const bool lo = lane < 16;
    const int l16 = lane & 15;
    const int j = n0 + l16;                            // hidden column (B-matrix col)
    const int koff = lo ? 0 : 2;                       // f32 A/B layout: lo lanes K{0,1}, hi K{2,3}

    const float* arow = xg_in + (m0 + l16) * XW;
    const float* wi0 = w_ih + (long)(0 * HH + j) * EE;
    const float* wi1 = w_ih + (long)(1 * HH + j) * EE;
    const float* wi2 = w_ih + (long)(2 * HH + j) * EE;
    const float* wi3 = w_ih + (long)(3 * HH + j) * EE;
    const float* wh0 = w_hh + (long)(0 * HH + j) * HH;
    const float* wh1 = w_hh + (long)(1 * HH + j) * HH;
    const float* wh2 = w_hh + (long)(2 * HH + j) * HH;
    const float* wh3 = w_hh + (long)(3 * HH + j) * HH;

    v8f ai = {0.f,0.f,0.f,0.f,0.f,0.f,0.f,0.f};
    v8f af = ai, ag = ai, ao = ai;

    for (int k = 0; k < EE; k += 4) {
        v2f a = *(const v2f*)(arow + k + koff);
        v2f bi = *(const v2f*)(wi0 + k + koff);
        v2f bf = *(const v2f*)(wi1 + k + koff);
        v2f bg = *(const v2f*)(wi2 + k + koff);
        v2f bo = *(const v2f*)(wi3 + k + koff);
        ai = __builtin_amdgcn_wmma_f32_16x16x4_f32(false, a, false, bi, (short)0, ai, false, false);
        af = __builtin_amdgcn_wmma_f32_16x16x4_f32(false, a, false, bf, (short)0, af, false, false);
        ag = __builtin_amdgcn_wmma_f32_16x16x4_f32(false, a, false, bg, (short)0, ag, false, false);
        ao = __builtin_amdgcn_wmma_f32_16x16x4_f32(false, a, false, bo, (short)0, ao, false, false);
    }
    for (int k = 0; k < HH; k += 4) {
        v2f a = *(const v2f*)(arow + EE + k + koff);
        v2f bi = *(const v2f*)(wh0 + k + koff);
        v2f bf = *(const v2f*)(wh1 + k + koff);
        v2f bg = *(const v2f*)(wh2 + k + koff);
        v2f bo = *(const v2f*)(wh3 + k + koff);
        ai = __builtin_amdgcn_wmma_f32_16x16x4_f32(false, a, false, bi, (short)0, ai, false, false);
        af = __builtin_amdgcn_wmma_f32_16x16x4_f32(false, a, false, bf, (short)0, af, false, false);
        ag = __builtin_amdgcn_wmma_f32_16x16x4_f32(false, a, false, bg, (short)0, ag, false, false);
        ao = __builtin_amdgcn_wmma_f32_16x16x4_f32(false, a, false, bo, (short)0, ao, false, false);
    }

    const float bi_ = b_ih[0 * HH + j] + b_hh[0 * HH + j];
    const float bf_ = b_ih[1 * HH + j] + b_hh[1 * HH + j];
    const float bg_ = b_ih[2 * HH + j] + b_hh[2 * HH + j];
    const float bo_ = b_ih[3 * HH + j] + b_hh[3 * HH + j];

    #pragma unroll
    for (int r = 0; r < 8; ++r) {
        int m = m0 + (lo ? r : 8 + r);                 // C layout: VGPR r -> M=r / M=8+r
        float ig = 1.f / (1.f + __expf(-(ai[r] + bi_)));
        float fg = 1.f / (1.f + __expf(-(af[r] + bf_)));
        float gg = tanhf(ag[r] + bg_);
        float og = 1.f / (1.f + __expf(-(ao[r] + bo_)));
        float c  = fg * cbuf[m * HH + j] + ig * gg;
        float h  = og * tanhf(c);
        cbuf[m * HH + j]         = c;
        xg_out[m * XW + EE + j]  = h;
        hbf[m * HH + j]          = (__bf16)h;
    }
}

// ---------------- logits GEMM: bf16 WMMA, 2 vocab tiles per wave ----------------

__device__ __forceinline__ void logits_epilogue(const v8f& c0, const v8f& c1,
                                                const float* __restrict__ fc_b,
                                                float* __restrict__ out,
                                                int m0, int n0, int l16, bool lo, int t) {
    int n = n0 + l16;
    float bias0 = fc_b[n];
    float bias1 = fc_b[n + 16];
    #pragma unroll
    for (int r = 0; r < 8; ++r) {
        int mm = m0 + (lo ? r : 8 + r);
        long base = (long)mm * ((long)TT * VV) + (long)t * VV;
        out[base + n]      = c0[r] + bias0;
        out[base + n + 16] = c1[r] + bias1;
    }
}

// Preconverted-bf16 weights path. Grid 500 blocks x 8 waves; wave = 1 batch tile x 2 vocab tiles.
__global__ __launch_bounds__(256) void logits_kernel_bf16(
    const __bf16* __restrict__ h_bf,  // [BB][HH]
    const __bf16* __restrict__ w_bf,  // [VV][HH]
    const float* __restrict__ fc_b,
    float* __restrict__ out, int t)
{
    const int lane = threadIdx.x & 31;
    const int wave = threadIdx.x >> 5;
    const int m0 = (wave & 3) * 16;
    const int n0 = blockIdx.x * 64 + (wave >> 2) * 32;
    const bool lo = lane < 16;
    const int l16 = lane & 15;
    const __bf16* arow = h_bf + (m0 + l16) * HH;
    const __bf16* br0  = w_bf + (long)(n0 + l16) * HH;
    const __bf16* br1  = w_bf + (long)(n0 + 16 + l16) * HH;

    v8f c0 = {0.f,0.f,0.f,0.f,0.f,0.f,0.f,0.f};
    v8f c1 = c0;
    for (int k0 = 0; k0 < HH; k0 += 32) {
        union { v16bf v; uint4 q[2]; } a, b0, b1;
        int ka = k0 + (lo ? 0 : 16);     // A 16x32 bf16: lo K0..15, hi K16..31
        a.q[0] = *(const uint4*)(arow + ka);
        a.q[1] = *(const uint4*)(arow + ka + 8);
        int kb = k0 + (lo ? 0 : 16);     // B 32x16 bf16: lo K0..15, hi K16..31
        b0.q[0] = *(const uint4*)(br0 + kb);
        b0.q[1] = *(const uint4*)(br0 + kb + 8);
        b1.q[0] = *(const uint4*)(br1 + kb);
        b1.q[1] = *(const uint4*)(br1 + kb + 8);
        c0 = __builtin_amdgcn_wmma_f32_16x16x32_bf16(false, a.v, false, b0.v, (short)0, c0, false, false);
        c1 = __builtin_amdgcn_wmma_f32_16x16x32_bf16(false, a.v, false, b1.v, (short)0, c1, false, false);
    }
    logits_epilogue(c0, c1, fc_b, out, m0, n0, l16, lo, t);
}

// Fallback: load fc_w as f32 and convert in-register (when ws too small to preconvert).
__global__ __launch_bounds__(256) void logits_kernel_f32(
    const __bf16* __restrict__ h_bf,
    const float* __restrict__ fc_w,   // [VV][HH] f32
    const float* __restrict__ fc_b,
    float* __restrict__ out, int t)
{
    const int lane = threadIdx.x & 31;
    const int wave = threadIdx.x >> 5;
    const int m0 = (wave & 3) * 16;
    const int n0 = blockIdx.x * 64 + (wave >> 2) * 32;
    const bool lo = lane < 16;
    const int l16 = lane & 15;
    const __bf16* arow = h_bf + (m0 + l16) * HH;
    const float* br0 = fc_w + (long)(n0 + l16) * HH;
    const float* br1 = fc_w + (long)(n0 + 16 + l16) * HH;

    v8f c0 = {0.f,0.f,0.f,0.f,0.f,0.f,0.f,0.f};
    v8f c1 = c0;
    for (int k0 = 0; k0 < HH; k0 += 32) {
        union { v16bf v; uint4 q[2]; __bf16 h[16]; } a, b0, b1;
        int ka = k0 + (lo ? 0 : 16);
        a.q[0] = *(const uint4*)(arow + ka);
        a.q[1] = *(const uint4*)(arow + ka + 8);
        int kb = k0 + (lo ? 0 : 16);
        #pragma unroll
        for (int q = 0; q < 4; ++q) {
            float4 f = *(const float4*)(br0 + kb + q * 4);
            b0.h[q*4+0] = (__bf16)f.x; b0.h[q*4+1] = (__bf16)f.y;
            b0.h[q*4+2] = (__bf16)f.z; b0.h[q*4+3] = (__bf16)f.w;
            float4 g = *(const float4*)(br1 + kb + q * 4);
            b1.h[q*4+0] = (__bf16)g.x; b1.h[q*4+1] = (__bf16)g.y;
            b1.h[q*4+2] = (__bf16)g.z; b1.h[q*4+3] = (__bf16)g.w;
        }
        c0 = __builtin_amdgcn_wmma_f32_16x16x32_bf16(false, a.v, false, b0.v, (short)0, c0, false, false);
        c1 = __builtin_amdgcn_wmma_f32_16x16x32_bf16(false, a.v, false, b1.v, (short)0, c1, false, false);
    }
    logits_epilogue(c0, c1, fc_b, out, m0, n0, l16, lo, t);
}

// ---------------- argmax + teacher forcing ----------------

__global__ __launch_bounds__(256) void argmax_kernel(
    const float* __restrict__ out, const int* __restrict__ target,
    const int* __restrict__ tf_mask, int t, int* __restrict__ tokens)
{
    __shared__ float sval[256];
    __shared__ int   sidx[256];
    const int b = blockIdx.x;
    const float* row = out + (long)b * TT * VV + (long)t * VV;
    float best = -INFINITY; int bi = 0;
    for (int v = threadIdx.x; v < VV; v += 256) {
        float x = row[v];
        if (x > best) { best = x; bi = v; }
    }
    sval[threadIdx.x] = best; sidx[threadIdx.x] = bi;
    __syncthreads();
    for (int s = 128; s > 0; s >>= 1) {
        if (threadIdx.x < s) {
            float ov = sval[threadIdx.x + s]; int oi = sidx[threadIdx.x + s];
            if (ov > sval[threadIdx.x] ||
                (ov == sval[threadIdx.x] && oi < sidx[threadIdx.x])) {
                sval[threadIdx.x] = ov; sidx[threadIdx.x] = oi;
            }
        }
        __syncthreads();
    }
    if (threadIdx.x == 0)
        tokens[b] = (tf_mask[t] > 0) ? target[b * TT + t] : sidx[0];
}

__global__ void final_kernel(const float* __restrict__ xg0, const float* __restrict__ cbuf,
                             float* __restrict__ out) {
    int i = blockIdx.x * blockDim.x + threadIdx.x;
    if (i < BB * HH) {
        int b = i / HH, j = i % HH;
        out[(long)BB * TT * VV + i]           = xg0[b * XW + EE + j];
        out[(long)BB * TT * VV + BB * HH + i] = cbuf[i];
    }
}

// ---------------- host ----------------

extern "C" void kernel_launch(void* const* d_in, const int* in_sizes, int n_in,
                              void* d_out, int out_size, void* d_ws, size_t ws_size,
                              hipStream_t stream)
{
    (void)in_sizes; (void)n_in; (void)out_size;
    const float* enc_h  = (const float*)d_in[0];
    const float* enc_c  = (const float*)d_in[1];
    const int*   target = (const int*)d_in[2];
    const int*   tfmask = (const int*)d_in[3];
    const float* emb    = (const float*)d_in[4];
    const float* w_ih   = (const float*)d_in[5];
    const float* w_hh   = (const float*)d_in[6];
    const float* b_ih   = (const float*)d_in[7];
    const float* b_hh   = (const float*)d_in[8];
    const float* fc_w   = (const float*)d_in[9];
    const float* fc_b   = (const float*)d_in[10];
    float* out = (float*)d_out;

    char* ws = (char*)d_ws;
    size_t o = 0;
    float* xg   = (float*)(ws + o); o += (size_t)2 * BB * XW * 4;  // ping-pong [x|h]
    float* cbuf = (float*)(ws + o); o += (size_t)BB * HH * 4;
    __bf16* hbf = (__bf16*)(ws + o); o += (size_t)BB * HH * 2;
    int* tokens = (int*)(ws + o);   o += 256;
    __bf16* wbf = (__bf16*)(ws + o);
    const size_t need_pre = o + (size_t)VV * HH * 2;
    const bool preconv = (ws_size >= need_pre);   // ws_size is launch-constant -> deterministic

    if (preconv)
        convert_fcw_kernel<<<(VV * HH) / (256 * 8), 256, 0, stream>>>(fc_w, wbf);

    init_kernel<<<(BB * HH + 255) / 256, 256, 0, stream>>>(enc_h, enc_c, xg, cbuf, tokens);

    for (int t = 0; t < TT; ++t) {
        float* xg_in  = xg + (t & 1) * BB * XW;
        float* xg_out = xg + ((t + 1) & 1) * BB * XW;
        embed_kernel<<<BB, 256, 0, stream>>>(emb, tokens, xg_in);
        cell_wmma_kernel<<<16, 256, 0, stream>>>(xg_in, w_ih, w_hh, b_ih, b_hh,
                                                 cbuf, xg_out, hbf);
        if (preconv)
            logits_kernel_bf16<<<VV / 64, 256, 0, stream>>>(hbf, wbf, fc_b, out, t);
        else
            logits_kernel_f32<<<VV / 64, 256, 0, stream>>>(hbf, fc_w, fc_b, out, t);
        if (t < TT - 1)
            argmax_kernel<<<BB, 256, 0, stream>>>(out, target, tfmask, t, tokens);
    }
    final_kernel<<<(BB * HH + 255) / 256, 256, 0, stream>>>(xg, cbuf, out);
}
